// PatchQuantumGenerator_11106785427531
// MI455X (gfx1250) — compile-verified, matrix operand order checked by reading, not run on hardware
//
#include <hip/hip_runtime.h>
#include <hip/hip_bf16.h>

// ---------------------------------------------------------------------------
// PatchQuantumGenerator on MI455X (gfx1250, wave32, WMMA + async-to-LDS)
//
// out[b, g*256+n] = s_n^2 / max_n(s_n^2),  s = (first 256 rows of M_g) @ v_b
//   v_b   : product state from first RY layer (x angles)       -> f16 A-frags
//   M_g   : 6x (RY Kronecker layer + CZ sign), orthogonal      -> f16 B-frags
// sum(probs) normalization cancels against max-normalization exactly.
//
// Workspace layout (needs 16 MiB):
//   Vfrag: 4096x1024 f16, stored as WMMA A-fragments  (8 MiB)
//   Wfrag: 16 x 1024x256 f16, stored as WMMA B-fragments (8 MiB)
// ---------------------------------------------------------------------------

typedef _Float16 half16 __attribute__((ext_vector_type(16)));
typedef float    floatx8 __attribute__((ext_vector_type(8)));

#define NQ 10
#define DIM 1024
#define PATCH 256
#define GCOUNT 16
#define BATCH 4096
#define QDEPTH 6

// elements per fragment (32 lanes x 16 f16) = one 16x32 A tile or 32x16 B tile
#define FRAG_ELEMS 512
// f16 elements of Wfrag per generator: 32 k-tiles * 16 n-tiles * 512
#define W_PER_G (32 * 16 * FRAG_ELEMS)
// bytes of one K-step of B fragments for a whole block (16 n-tiles)
#define BSTEP_BYTES (16 * FRAG_ELEMS * 2)   // 16 KiB, contiguous in Wfrag

// ---------------------------------------------------------------------------
// Kernel 1: build V = first-layer product states, directly in WMMA A-fragment
// layout.  A 16x32 f16 tile: lanes 0-15 hold row M=lane, K in {0..7,16..23};
// lanes 16-31 hold row M=lane-16, K in {8..15,24..31}; 16 f16 per lane.
// One wave per fragment; 8192 fragments total (256 m-tiles x 32 k-tiles).
// ---------------------------------------------------------------------------
__global__ __launch_bounds__(256) void make_V(const float* __restrict__ x,
                                              _Float16* __restrict__ Vfrag) {
  const int tid  = threadIdx.x;
  const int lane = tid & 31;
  const int F    = blockIdx.x * 8 + (tid >> 5);   // fragment id, 0..8191
  const int mt   = F >> 5;                        // m-tile 0..255
  const int kt   = F & 31;                        // k-tile 0..31

  const int m = lane & 15;
  const int b = mt * 16 + m;
  const float* xb = x + b * NQ;

  float c[NQ], s[NQ];
#pragma unroll
  for (int w = 0; w < NQ; ++w) {
    float h = 0.5f * xb[w];
    c[w] = __cosf(h);
    s[w] = __sinf(h);
  }

  // K indices this lane owns: chunk base (lane>=16 ? 8 : 0), chunks {+0,+16}
  const int kbase = kt * 32 + ((lane >> 4) << 3);
  half16 v;
#pragma unroll
  for (int e = 0; e < 16; ++e) {
    const int j = kbase + (e < 8 ? e : e + 8);  // e>=8 -> +16 + (e-8)
    float p = 1.0f;
#pragma unroll
    for (int w = 0; w < NQ; ++w)
      p *= ((j >> (9 - w)) & 1) ? s[w] : c[w];  // RY column 0 of |0> : (c, s)
    v[e] = (_Float16)p;
  }
  *(half16*)(Vfrag + (size_t)F * FRAG_ELEMS + lane * 16) = v;
}

// ---------------------------------------------------------------------------
// Kernel 2: build W_g = first 256 rows of M_g, one wave32 per basis column j.
// The 1024-amplitude state lives in 32 VGPRs per lane: amplitude index
// i = r*32 + lane.  Pair-stride >= 32 -> register butterfly; < 32 -> shfl_xor.
// First per-g layer applied to e_j is still a product state -> direct init.
// ---------------------------------------------------------------------------
__global__ __launch_bounds__(256) void make_W(const float* __restrict__ qp,
                                              _Float16* __restrict__ Wfrag) {
  const int tid    = threadIdx.x;
  const int lane   = tid & 31;
  const int waveId = blockIdx.x * (blockDim.x >> 5) + (tid >> 5);
  const int g = waveId >> 10;        // 0..15
  const int j = waveId & 1023;       // column of M_g

  const float* qg = qp + g * (QDEPTH * NQ);
  float st[32];

  // ---- depth 0: column of a Kronecker-RY layer is a product over qubits ----
  {
    float c[NQ], s[NQ];
#pragma unroll
    for (int w = 0; w < NQ; ++w) {
      float h = 0.5f * qg[w];
      c[w] = __cosf(h);
      s[w] = __sinf(h);
    }
#pragma unroll
    for (int r = 0; r < 32; ++r) {
      const int i = r * 32 + lane;
      float v = 1.0f;
#pragma unroll
      for (int w = 0; w < NQ; ++w) {
        const int bi = (i >> (9 - w)) & 1;
        const int bj = (j >> (9 - w)) & 1;
        // RY = [[c,-s],[s,c]] entry [bi][bj]
        v *= (bi == bj) ? c[w] : (bi ? s[w] : -s[w]);
      }
      // CZ sign: parity of adjacent-bit ANDs
      st[r] = (__popc(i & (i >> 1)) & 1) ? -v : v;
    }
  }

  // ---- depths 1..5: in-register statevector butterflies ----
  for (int d = 1; d < QDEPTH; ++d) {
    const float* qd = qg + d * NQ;
#pragma unroll
    for (int w = 0; w < NQ; ++w) {
      const float h = 0.5f * qd[w];
      const float c = __cosf(h), s = __sinf(h);
      const int p = 9 - w;  // pair-stride bit
      if (p >= 5) {
        const int msk = 1 << (p - 5);
#pragma unroll
        for (int r = 0; r < 32; ++r) {
          if (!(r & msk)) {
            const float a0 = st[r], a1 = st[r | msk];
            st[r]       = c * a0 - s * a1;
            st[r | msk] = s * a0 + c * a1;
          }
        }
      } else {
        const int lmask = 1 << p;
        const int hi = (lane >> p) & 1;
#pragma unroll
        for (int r = 0; r < 32; ++r) {
          const float mine  = st[r];
          const float other = __shfl_xor(mine, lmask, 32);
          st[r] = hi ? (s * other + c * mine) : (c * mine - s * other);
        }
      }
    }
#pragma unroll
    for (int r = 0; r < 32; ++r) {
      const int i = r * 32 + lane;
      if (__popc(i & (i >> 1)) & 1) st[r] = -st[r];
    }
  }

  // ---- scatter first 256 amplitudes (i<256) into WMMA B-fragment layout ----
  // B 32x16 f16 tile: lane n holds col n, K in {0..7,16..23}; lane n+16 holds
  // col n, K in {8..15,24..31}.  Here K == j (column index), N == i.
  const int kt  = j >> 5;
  const int grp = (j >> 3) & 1;
  const int e   = (j & 7) + (((j >> 4) & 1) << 3);
  _Float16* Wg = Wfrag + (size_t)g * W_PER_G;
#pragma unroll
  for (int r = 0; r < 8; ++r) {
    const int i     = r * 32 + lane;       // amplitude index < 256
    const int nt    = i >> 4;
    const int lanep = (i & 15) + 16 * grp;
    Wg[(size_t)(kt * 16 + nt) * FRAG_ELEMS + lanep * 16 + e] = (_Float16)st[r];
  }
}

// ---------------------------------------------------------------------------
// Kernel 3: out(32x256 per block) = V(32x1024) @ W_g(1024x256), squared,
// row-normalized by its max.
//
// B path is the CDNA5 async pipeline: per K-step the 16 KiB of B fragments
// (contiguous in Wfrag) are pulled into a double-buffered LDS stage with
// global_load_async_to_lds_b128 (ASYNCcnt), then consumed with ds_load_b128.
// The LDS stage is shared by all 8 waves, halving global B traffic and taking
// the copy latency off the wmma dependency chain (s_wait_asynccnt <= 4 keeps
// the next step's copy in flight while this step computes).
// A fragments (1 KiB, shared/broadcast) stay as direct global loads.
// ---------------------------------------------------------------------------
struct SMem {
  union {
    _Float16 bstage[2][16 * FRAG_ELEMS];  // 2 x 16 KiB double buffer
    float    sq[32][257];                 // epilogue scratch (after K loop)
  };
};

__global__ __launch_bounds__(256) void qgemm(const _Float16* __restrict__ Vfrag,
                                             const _Float16* __restrict__ Wfrag,
                                             float* __restrict__ out) {
  __shared__ SMem sm;

  const int tid  = threadIdx.x;
  const int lane = tid & 31;
  const int w    = tid >> 5;
  const int g    = blockIdx.y;        // generator
  const int mb   = blockIdx.x;        // 128 blocks of M=32
  const int mtl  = w & 1;             // local m-tile
  const int nh   = w >> 1;            // n-quarter (64 cols)
  const int mt   = mb * 2 + mtl;      // global m-tile 0..255

  floatx8 acc[4] = {};

  const _Float16* Abase = Vfrag + (size_t)mt * 32 * FRAG_ELEMS + lane * 16;
  const _Float16* Bg    = Wfrag + (size_t)g * W_PER_G;

  // per-thread addresses for the async stage: 256 thr x 16 B x 4 issues = 16KiB
  const unsigned voff = (unsigned)tid * 16u;  // global byte offset within step
  const unsigned lds0 =
      (unsigned)(uintptr_t)(&sm.bstage[0][0]) + (unsigned)tid * 16u;
  const unsigned lds1 =
      (unsigned)(uintptr_t)(&sm.bstage[1][0]) + (unsigned)tid * 16u;

#define B_ASYNC_STEP(ldsd, sbase)                                           \
  asm volatile(                                                             \
      "global_load_async_to_lds_b128 %0, %1, %2 offset:0\n\t"               \
      "global_load_async_to_lds_b128 %0, %1, %2 offset:4096\n\t"            \
      "global_load_async_to_lds_b128 %0, %1, %2 offset:8192\n\t"            \
      "global_load_async_to_lds_b128 %0, %1, %2 offset:12288"               \
      :: "v"(ldsd), "v"(voff), "s"(sbase) : "memory")

  // prologue: stage K-step 0 into buffer 0
  B_ASYNC_STEP(lds0, Bg);

  for (int kt = 0; kt < 32; ++kt) {
    // kick off next K-step copy into the other buffer
    if (kt + 1 < 32) {
      const unsigned ldsn = ((kt + 1) & 1) ? lds1 : lds0;
      B_ASYNC_STEP(ldsn, Bg + (size_t)(kt + 1) * (16 * FRAG_ELEMS));
      asm volatile("s_wait_asynccnt 0x4" ::: "memory");  // current step done
    } else {
      asm volatile("s_wait_asynccnt 0x0" ::: "memory");
    }
    __syncthreads();  // buffer (kt&1) fully written by all waves

    const half16 a = *(const half16*)(Abase + (size_t)kt * FRAG_ELEMS);
    if (kt + 1 < 32)  // hint next A fragment into cache (global_prefetch)
      __builtin_prefetch(Abase + (size_t)(kt + 1) * FRAG_ELEMS, 0, 1);

    const _Float16* Bl = &sm.bstage[kt & 1][0] + lane * 16;
#pragma unroll
    for (int n = 0; n < 4; ++n) {
      const half16 b = *(const half16*)(Bl + (size_t)(nh * 4 + n) * FRAG_ELEMS);
      acc[n] = __builtin_amdgcn_wmma_f32_16x16x32_f16(
          false, a, false, b, (short)0, acc[n], false, false);
    }
    __syncthreads();  // all waves done reading before this buffer is reused
  }
#undef B_ASYNC_STEP

  // C layout: VGPR r -> row (r + 8*(lane>>4)) of the tile, col = lane&15.
  // (bstage is dead here; sq overlays it in the union, guarded by the final
  //  __syncthreads() of the K loop.)
  const int rowbase = mtl * 16 + 8 * (lane >> 4);
  const int nlo     = lane & 15;
#pragma unroll
  for (int n = 0; n < 4; ++n) {
#pragma unroll
    for (int r = 0; r < 8; ++r) {
      const float v = acc[n][r];
      sm.sq[rowbase + r][nh * 64 + n * 16 + nlo] = v * v;
    }
  }
  __syncthreads();

  // per-row max over 256 cols: 8 threads per row, shfl-combine, normalize
  const int row = tid >> 3;   // 0..31
  const int oct = tid & 7;    // 32-col slice
  float mx = 0.0f;
#pragma unroll 4
  for (int c2 = 0; c2 < 32; ++c2) mx = fmaxf(mx, sm.sq[row][oct * 32 + c2]);
  mx = fmaxf(mx, __shfl_xor(mx, 1, 32));
  mx = fmaxf(mx, __shfl_xor(mx, 2, 32));
  mx = fmaxf(mx, __shfl_xor(mx, 4, 32));
  const float inv = 1.0f / mx;

  const int b = mb * 32 + row;
  float* orow = out + (size_t)b * (GCOUNT * PATCH) + g * PATCH;
#pragma unroll 4
  for (int c2 = 0; c2 < 32; ++c2) {
    const int col = oct * 32 + c2;
    orow[col] = sm.sq[row][col] * inv;
  }
}

// ---------------------------------------------------------------------------
extern "C" void kernel_launch(void* const* d_in, const int* in_sizes, int n_in,
                              void* d_out, int out_size, void* d_ws,
                              size_t ws_size, hipStream_t stream) {
  (void)in_sizes; (void)n_in; (void)out_size; (void)ws_size;
  const float* x  = (const float*)d_in[0];   // 4096 x 10
  const float* qp = (const float*)d_in[1];   // 16 x 60
  float* out = (float*)d_out;                // 4096 x 4096 fp32

  _Float16* Vfrag = (_Float16*)d_ws;                       // 8 MiB
  _Float16* Wfrag = Vfrag + (size_t)BATCH * DIM;           // 8 MiB

  make_V<<<1024, 256, 0, stream>>>(x, Vfrag);              // 8192 frags
  make_W<<<2048, 256, 0, stream>>>(qp, Wfrag);             // 16384 column sims
  qgemm<<<dim3(128, GCOUNT), 256, 0, stream>>>(Vfrag, Wfrag, out);
}
